// AttentionWithContext_54769422958918
// MI455X (gfx1250) — compile-verified, matrix operand order checked by reading, not run on hardware
//
#include <hip/hip_runtime.h>
#include <hip/hip_bf16.h>

// Problem constants (match reference setup_inputs)
#define B_    32
#define T_    4096
#define C_    512
#define TTILE 128
#define NCB   (C_ / 16)          // 32 column blocks
#define PSTR  (C_ + 8)           // padded LDS panel row stride (bf16 elems) -> 1040 B
#define EPSF  1.1920928955078125e-07f

typedef __attribute__((ext_vector_type(16))) __bf16       bfx16;
typedef __attribute__((ext_vector_type(8)))  __bf16       bfx8;
typedef __attribute__((ext_vector_type(8)))  float        fx8;
typedef __attribute__((ext_vector_type(4)))  unsigned int ui32x4;
typedef __attribute__((ext_vector_type(8)))  int          i32x8;
typedef __attribute__((ext_vector_type(4)))  int          i32x4;

__device__ __forceinline__ unsigned short f32_to_bf16_rne(float f) {
  unsigned int u = __float_as_uint(f);
  unsigned int lsb = (u >> 16) & 1u;
  u += 0x7fffu + lsb;               // round-to-nearest-even
  return (unsigned short)(u >> 16);
}

// ---------------------------------------------------------------------------
// TDM: DMA one 16-row x 512-col bf16 panel of Wt into LDS, inserting 16 B of
// padding after every 1024 B row (pad_interval=256 DWORDs, pad_amount=4 DWORDs)
// so the LDS panel row stride is PSTR=520 bf16 (bank-conflict-free b128 reads).
// D# layout per CDNA5 ISA ch.8 (groups 0/1; 2D tensor so groups 2/3 zero).
// This toolchain exposes the 6-arg builtin: (g0, g1, g2, g3, g4, cpol).
// ---------------------------------------------------------------------------
__device__ __forceinline__ void tdm_load_panel(const unsigned short* gsrc,
                                               unsigned int lds_byte_addr) {
  unsigned long long ga = (unsigned long long)(uintptr_t)gsrc;
  ui32x4 g0;
  g0[0] = 1u;                                              // count=1 (valid), user mode
  g0[1] = lds_byte_addr;                                   // lds_addr [63:32]
  g0[2] = (unsigned int)(ga & 0xFFFFFFFFull);              // global_addr lo
  g0[3] = (unsigned int)((ga >> 32) & 0x1FFFFFFull)        // global_addr [56:32]
          | (2u << 30);                                    // type=2 ("image")
  i32x8 g1;
  g1[0] = (int)0x07D10000u;          // data_size=2B | pad_enable | pad_interval=7 | pad_amount=3
  g1[1] = (int)(((unsigned)C_ & 0xFFFFu) << 16);           // tensor_dim0 lo16
  g1[2] = (int)(((unsigned)C_ & 0xFFFFu) << 16);           // tensor_dim1 lo16
  g1[3] = (int)(((unsigned)C_ & 0xFFFFu) << 16);           // tile_dim0 = 512
  g1[4] = 16;                                              // tile_dim1 = 16 rows
  g1[5] = C_;                                              // tensor_dim0_stride = 512 elems
  g1[6] = 0;
  g1[7] = 0;
  i32x4 z4 = {0, 0, 0, 0};
  i32x8 z8 = {0, 0, 0, 0, 0, 0, 0, 0};
  __builtin_amdgcn_tensor_load_to_lds(g0, g1, z4, z4, z8, 0);
}

// ---------------------------------------------------------------------------
// Kernel 0: W[k][d] (f32) -> Wt[d][k] (bf16), zero accumulators
// ---------------------------------------------------------------------------
__global__ __launch_bounds__(256) void awc_prep(const float* __restrict__ W,
                                                unsigned short* __restrict__ Wt,
                                                float* __restrict__ outacc,
                                                float* __restrict__ denom) {
  int idx = blockIdx.x * blockDim.x + threadIdx.x;
  if (idx < C_ * C_) {
    int d = idx % C_;
    int k = idx / C_;
    Wt[(size_t)d * C_ + k] = f32_to_bf16_rne(W[(size_t)k * C_ + d]);
  }
  if (idx < B_ * C_) outacc[idx] = 0.0f;
  if (idx < B_)      denom[idx]  = 0.0f;
}

// ---------------------------------------------------------------------------
// Kernel 1: per (batch, 128-row T-tile):
//   s[t] = exp( u . tanh( x[t,:] @ W + b ) )   via bf16 WMMA, f32 accum
//   W column panels DMA'd to LDS by the Tensor Data Mover, double-buffered.
//   atomic: denom[b] += sum_t s[t];  outacc[b,:] += sum_t x[t,:] * s[t]
// ---------------------------------------------------------------------------
__global__ __launch_bounds__(256) void awc_main(const float* __restrict__ x,
                                                const unsigned short* __restrict__ Wt,
                                                const float* __restrict__ bvec,
                                                const float* __restrict__ uvec,
                                                float* __restrict__ outacc,
                                                float* __restrict__ denom) {
  extern __shared__ char smem[];
  unsigned short* wpan = (unsigned short*)smem;                              // [2][16][PSTR]
  unsigned short* xs   = (unsigned short*)(smem + 2 * 16 * PSTR * 2);        // [TTILE][C_]
  float*          earr = (float*)(smem + 2 * 16 * PSTR * 2
                                       + (size_t)TTILE * C_ * 2);            // [TTILE]

  const int tid  = threadIdx.x;
  const int lane = tid & 31;
  const int wid  = tid >> 5;                    // 0..7 waves
  const int bidx = blockIdx.x / (T_ / TTILE);
  const int t0   = (blockIdx.x % (T_ / TTILE)) * TTILE;

  const float* xg = x + ((size_t)bidx * T_ + t0) * C_;

  // Kick off the first W panel DMA (wave 0 only; TDM ignores EXEC).
  if (wid == 0)
    tdm_load_panel(Wt, (unsigned int)(uintptr_t)&wpan[0]);

  // Stage x tile as bf16 in LDS (coalesced, overlaps with the TDM transfer)
  for (int i = tid; i < TTILE * C_; i += 256)
    xs[i] = f32_to_bf16_rne(xg[i]);

  const int hi  = lane >> 4;                    // K-half select (ISA 16-bit A/B layout)
  const int l16 = lane & 15;
  const int m0  = wid * 16;                     // wave's row base within tile

  const int row = m0 + l16;                     // A-matrix: lanes 0-15 hold rows M=0..15
  const unsigned short* xrow = &xs[(size_t)row * C_];

  float ait[8];
#pragma unroll
  for (int r = 0; r < 8; ++r) ait[r] = 0.0f;

  for (int cb = 0; cb < NCB; ++cb) {
    const int buf = cb & 1;
    if (wid == 0) {
      if (cb + 1 < NCB) {
        // Prefetch next panel into the other buffer, then wait for current one.
        tdm_load_panel(Wt + (size_t)(cb + 1) * 16 * C_,
                       (unsigned int)(uintptr_t)&wpan[(buf ^ 1) * 16 * PSTR]);
        __builtin_amdgcn_s_wait_tensorcnt(1);   // TENSORcnt in-order: current done
      } else {
        __builtin_amdgcn_s_wait_tensorcnt(0);
      }
    }
    __syncthreads();                            // panel visible to all waves

    // B-matrix: lane holds column N = l16 of this block, from padded LDS panel
    const unsigned short* wcol = &wpan[buf * 16 * PSTR + (size_t)l16 * PSTR];

    fx8 acc = {0.f, 0.f, 0.f, 0.f, 0.f, 0.f, 0.f, 0.f};
#pragma unroll
    for (int kb = 0; kb < C_; kb += 32) {
      // 16-bit A 16x32 layout: VGPR0-3 = K[hi*8 .. hi*8+7], VGPR4-7 = K[16+hi*8 ..]
      bfx8 alo = *(const bfx8*)&xrow[kb + hi * 8];
      bfx8 ahi = *(const bfx8*)&xrow[kb + 16 + hi * 8];
      bfx16 afrag = __builtin_shufflevector(alo, ahi, 0, 1, 2, 3, 4, 5, 6, 7,
                                            8, 9, 10, 11, 12, 13, 14, 15);
      bfx8 blo = *(const bfx8*)&wcol[kb + hi * 8];
      bfx8 bhi = *(const bfx8*)&wcol[kb + 16 + hi * 8];
      bfx16 bfrag = __builtin_shufflevector(blo, bhi, 0, 1, 2, 3, 4, 5, 6, 7,
                                            8, 9, 10, 11, 12, 13, 14, 15);
      acc = __builtin_amdgcn_wmma_f32_16x16x32_bf16(
          /*neg_a=*/false, afrag, /*neg_b=*/false, bfrag,
          /*c_mod=*/(short)0, acc, /*reuse_a=*/false, /*reuse_b=*/false);
    }
    // C/D layout: acc[r] @ lane = element (row r + 8*hi, col cb*16 + l16)
    const int col = cb * 16 + l16;
    const float bb = bvec[col];
    const float uu = uvec[col];
#pragma unroll
    for (int r = 0; r < 8; ++r)
      ait[r] += tanhf(acc[r] + bb) * uu;

    __syncthreads();                            // all reads done before panel reuse
  }

  // Reduce over the 16 lanes of each half (same row, different cols)
#pragma unroll
  for (int r = 0; r < 8; ++r) {
    float v = ait[r];
    v += __shfl_xor(v, 1);
    v += __shfl_xor(v, 2);
    v += __shfl_xor(v, 4);
    v += __shfl_xor(v, 8);
    ait[r] = v;
  }

  if (l16 == 0) {
    float psum = 0.0f;
#pragma unroll
    for (int r = 0; r < 8; ++r) {
      float e = __expf(ait[r]);                 // reference: exp without max-subtract
      earr[m0 + 8 * hi + r] = e;
      psum += e;
    }
    atomicAdd(&denom[bidx], psum);
  }
  __syncthreads();

  // Weighted sum over the tile's rows: fp32 x (L2-resident tile), LDS-broadcast weights
  float* oacc = outacc + (size_t)bidx * C_;
  for (int c = tid; c < C_; c += 256) {
    float s = 0.0f;
#pragma unroll 4
    for (int t = 0; t < TTILE; ++t)
      s += xg[(size_t)t * C_ + c] * earr[t];
    atomicAdd(&oacc[c], s);
  }
}

// ---------------------------------------------------------------------------
// Kernel 2: out[b,c] = outacc[b,c] / (denom[b] + EPS)
// ---------------------------------------------------------------------------
__global__ __launch_bounds__(256) void awc_finalize(const float* __restrict__ outacc,
                                                    const float* __restrict__ denom,
                                                    float* __restrict__ out) {
  int i = blockIdx.x * blockDim.x + threadIdx.x;
  if (i < B_ * C_) out[i] = outacc[i] / (denom[i / C_] + EPSF);
}

// ---------------------------------------------------------------------------
extern "C" void kernel_launch(void* const* d_in, const int* in_sizes, int n_in,
                              void* d_out, int out_size, void* d_ws, size_t ws_size,
                              hipStream_t stream) {
  const float* x    = (const float*)d_in[0];   // [B,T,C]
  const float* W    = (const float*)d_in[1];   // [C,C]
  const float* bvec = (const float*)d_in[2];   // [C]
  const float* uvec = (const float*)d_in[3];   // [C]
  float*       out  = (float*)d_out;           // [B,C]

  // Workspace layout
  unsigned short* Wt     = (unsigned short*)d_ws;                          // 512 KB
  float*          outacc = (float*)((char*)d_ws + (size_t)C_ * C_ * 2);    // 64 KB
  float*          denom  = outacc + B_ * C_;                               // 128 B

  // 1) bf16-transpose W + zero accumulators
  awc_prep<<<(C_ * C_ + 255) / 256, 256, 0, stream>>>(W, Wt, outacc, denom);

  // 2) fused WMMA attention-score + weighted-sum pass (TDM-fed W panels)
  const int    nblocks = B_ * (T_ / TTILE);                   // 1024
  const size_t lds     = (size_t)2 * 16 * PSTR * 2            // W panel double buffer
                       + (size_t)TTILE * C_ * 2               // x tile (bf16)
                       + (size_t)TTILE * 4;                   // exp weights
  awc_main<<<nblocks, 256, lds, stream>>>(x, Wt, bvec, uvec, outacc, denom);

  // 3) normalize
  awc_finalize<<<(B_ * C_ + 255) / 256, 256, 0, stream>>>(outacc, denom, out);
}
